// NetVladCNN_82549271429466
// MI455X (gfx1250) — matrix operand order repaired and checked by reading, not compiled, not run on hardware
//
#include <hip/hip_runtime.h>
#include <hip/hip_bf16.h>
#include <math.h>

// ---------------------------------------------------------------------------
// NetVLAD fused kernel for MI455X (gfx1250, wave32, WMMA bf16 path)
//
// D=256, K=64, N=512*512=262144.  X = 256MB fp32 read exactly once
// (memory floor ~11us @ 23.3 TB/s).  Both 8.6-GFLOP GEMMs run on
// v_wmma_f32_16x16x32_bf16.  All WMMA operands are built from 128-bit LDS
// loads (2 x ds_load_b128 per operand) by keeping both X and X^T tiles in
// LDS -- affordable only because CDNA5 gives 320KB LDS per WGP.
// ---------------------------------------------------------------------------

#define D_DIM 256
#define K_DIM 64
#define N_PIX (512 * 512)
#define NT 128                 // pixels per tile
#define NTILES (N_PIX / NT)    // 2048
#define NBLOCKS 512
#define NTHREADS 256           // 8 wave32

// padded LDS row strides (elements); all multiples of 8 so 128-bit loads stay
// 16B-aligned, padded to break bank conflicts on row-strided access
#define XS  264                // x_lds  [n][d]  row stride (d-minor)
#define TS  136                // xT_lds [d][n]  row stride (n-minor)
#define WS  264                // w_lds  [k][d]  row stride
#define ABS 136                // ab_lds [k][n]  row stride

typedef __attribute__((ext_vector_type(16))) __bf16        v16bf;
typedef __attribute__((ext_vector_type(8)))  float         v8f;
typedef __attribute__((ext_vector_type(4)))  unsigned int  u32x4;

union U16BF { v16bf v; u32x4 q[2]; };

__device__ __forceinline__ unsigned short f2bf(float f) {
    unsigned u = __float_as_uint(f);
    unsigned r = u + 0x7FFFu + ((u >> 16) & 1u);   // round-to-nearest-even
    return (unsigned short)(r >> 16);
}
__device__ __forceinline__ unsigned pack2(float a, float b) {
    return (unsigned)f2bf(a) | ((unsigned)f2bf(b) << 16);
}
__device__ __forceinline__ u32x4 ld128(const unsigned short* p) {
    return *(const u32x4*)p;                       // ds_load_b128
}

// ---------------------------------------------------------------------------
// Kernel 0: zero the V / a_sum accumulators in workspace
// ---------------------------------------------------------------------------
__global__ void nv_zero_kernel(float* ws, int n) {
    int i = blockIdx.x * blockDim.x + threadIdx.x;
    if (i < n) ws[i] = 0.0f;
}

// ---------------------------------------------------------------------------
// Kernel 1: fused  logits -> softmax -> VLAD accumulation
// LDS carve (dynamic, ~217KB of the 320KB CDNA5 WGP budget):
//   x_lds  : bf16 [NT][XS]    67584 B   GEMM1-B (d-minor)
//   xT_lds : bf16 [D][TS]     69632 B   GEMM2-B (n-minor)
//   w_lds  : bf16 [K][WS]     33792 B   GEMM1-A
//   lg_lds : f32  [K][NT]     32768 B   logits / exp scratch
//   ab_lds : bf16 [K][ABS]    17408 B   GEMM2-A (softmax output)
//   b_lds, as_lds : f32 [K]     512 B
// ---------------------------------------------------------------------------
#define SM_X   0
#define SM_XT  67584
#define SM_W   137216
#define SM_LG  171008
#define SM_AB  203776
#define SM_B   221184
#define SM_AS  221440
#define SM_TOT 221696

__global__ void nv_fused_kernel(const float* __restrict__ xf,
                                const float* __restrict__ conv_w,
                                const float* __restrict__ conv_b,
                                float* __restrict__ Vg,
                                float* __restrict__ asumg) {
    extern __shared__ char smem[];
    unsigned short* x_lds  = (unsigned short*)(smem + SM_X);
    unsigned short* xT_lds = (unsigned short*)(smem + SM_XT);
    unsigned short* w_lds  = (unsigned short*)(smem + SM_W);
    float*          lg_lds = (float*)         (smem + SM_LG);
    unsigned short* ab_lds = (unsigned short*)(smem + SM_AB);
    float*          b_lds  = (float*)         (smem + SM_B);
    float*          as_lds = (float*)         (smem + SM_AS);

    unsigned* xt32 = (unsigned*)xT_lds;
    unsigned* w32  = (unsigned*)w_lds;

    const int tid  = threadIdx.x;
    const int wave = tid >> 5;
    const int lane = tid & 31;
    const int m    = lane & 15;    // row for A; column for B/C/D
    const int half = lane >> 4;

    // -------- block-invariant staging: conv_b, conv_w (f32 -> bf16) --------
    if (tid < K_DIM) { b_lds[tid] = conv_b[tid]; as_lds[tid] = 0.0f; }
    for (int i = tid; i < (K_DIM * D_DIM) / 4; i += NTHREADS) {
        float4 wv = ((const float4*)conv_w)[i];
        int k = (i * 4) >> 8, d = (i * 4) & 255;
        int e = k * WS + d;                         // d % 4 == 0 -> dword ok
        w32[e >> 1]       = pack2(wv.x, wv.y);
        w32[(e >> 1) + 1] = pack2(wv.z, wv.w);
    }

    // persistent V accumulators: wave owns 8 (kt x dtile) 16x16 f32 tiles
    const int kt = wave >> 1;           // 0..3  -> k rows kt*16..+15
    const int dh = (wave & 1) * 8;      // d-tile half: 0 or 8
    v8f vacc[8];
    const v8f vz = {};
#pragma unroll
    for (int j = 0; j < 8; ++j) vacc[j] = vz;

    for (int t = blockIdx.x; t < NTILES; t += gridDim.x) {
        const int n0 = t * NT;
        __syncthreads();   // protect x/xT/ab reuse from previous iteration

        {   // speculative prefetch of the next tile this block will touch
            int tn = t + gridDim.x;
            if (tn < NTILES)
                __builtin_prefetch(xf + (size_t)tid * N_PIX + (size_t)tn * NT, 0, 1);
        }

        // -------- stage X tile: coalesced float4 loads -> both LDS layouts -
        for (int j = 0; j < 32; ++j) {
            int l = j * NTHREADS + tid;             // 0..8191
            int d = l >> 5;                         // feature row
            int q = l & 31;                         // float4 index along n
            float4 xv = *(const float4*)(xf + (size_t)d * N_PIX + n0 + q * 4);
            int nb = q * 4;
            // transposed copy (strided u16 stores, pad breaks conflicts)
            x_lds[(nb + 0) * XS + d] = f2bf(xv.x);
            x_lds[(nb + 1) * XS + d] = f2bf(xv.y);
            x_lds[(nb + 2) * XS + d] = f2bf(xv.z);
            x_lds[(nb + 3) * XS + d] = f2bf(xv.w);
            // straight copy (contiguous dword stores)
            int e = d * TS + nb;
            xt32[e >> 1]       = pack2(xv.x, xv.y);
            xt32[(e >> 1) + 1] = pack2(xv.z, xv.w);
        }
        __syncthreads();

        // -------- GEMM1: logits(64x128) = Wconv(64x256) x X(256x128) -------
        // wave -> k-rows kt*16..+15, column half (wave&1)*64, 4 tiles of 16
        {
            v8f acc4[4] = {vz, vz, vz, vz};
#pragma unroll
            for (int dk = 0; dk < 8; ++dk) {        // D in chunks of 32
                U16BF a;                            // W tile 16x32 (A layout)
                const unsigned short* ap =
                    w_lds + (kt * 16 + m) * WS + dk * 32 + 8 * half;
                a.q[0] = ld128(ap);
                a.q[1] = ld128(ap + 16);
#pragma unroll
                for (int nt = 0; nt < 4; ++nt) {
                    const int ncol = (wave & 1) * 64 + nt * 16 + m;
                    U16BF b;                        // X chunk 32x16 (B layout)
                    const unsigned short* bp =
                        x_lds + ncol * XS + dk * 32 + 8 * half;
                    b.q[0] = ld128(bp);
                    b.q[1] = ld128(bp + 16);
                    acc4[nt] = __builtin_amdgcn_wmma_f32_16x16x32_bf16(
                        false, a.v, false, b.v, (short)0, acc4[nt], false, false);
                }
            }
#pragma unroll
            for (int nt = 0; nt < 4; ++nt) {
                const int nn = (wave & 1) * 64 + nt * 16 + m;
#pragma unroll
                for (int v = 0; v < 8; ++v)
                    lg_lds[(kt * 16 + v + 8 * half) * NT + nn] = acc4[nt][v];
            }
        }
        __syncthreads();

        // -------- softmax over K per column + a_sum accumulation -----------
        if (tid < NT) {
            const int n = tid;
            float mx = -3.4e38f;
            for (int k = 0; k < K_DIM; ++k)
                mx = fmaxf(mx, lg_lds[k * NT + n] + b_lds[k]);
            float s = 0.0f;
            for (int k = 0; k < K_DIM; ++k) {
                float e = __expf(lg_lds[k * NT + n] + b_lds[k] - mx);
                lg_lds[k * NT + n] = e;
                s += e;
            }
            float inv = 1.0f / s;
            for (int k = 0; k < K_DIM; ++k) {
                float a = lg_lds[k * NT + n] * inv;
                ab_lds[k * ABS + n] = f2bf(a);
                atomicAdd(&as_lds[k], a);           // ds_add_f32
            }
        }
        __syncthreads();

        // -------- GEMM2: V(64x256) += A(64x128) x X^T(128x256) -------------
        // wave -> k-rows kt*16..+15, d-tiles dh..dh+7; acc persistent in VGPRs
#pragma unroll
        for (int nk = 0; nk < 4; ++nk) {            // NT in chunks of 32
            U16BF a;                                // a_bar tile 16x32 (A)
            const unsigned short* ap =
                ab_lds + (kt * 16 + m) * ABS + nk * 32 + 8 * half;
            a.q[0] = ld128(ap);
            a.q[1] = ld128(ap + 16);
#pragma unroll
            for (int j = 0; j < 8; ++j) {
                const int dcol = (dh + j) * 16 + m;
                U16BF b;                            // X^T chunk 32x16 (B)
                const unsigned short* bp =
                    xT_lds + dcol * TS + nk * 32 + 8 * half;
                b.q[0] = ld128(bp);
                b.q[1] = ld128(bp + 16);
                vacc[j] = __builtin_amdgcn_wmma_f32_16x16x32_bf16(
                    false, a.v, false, b.v, (short)0, vacc[j], false, false);
            }
        }
    }
    __syncthreads();

    // -------- flush block partials: V and a_sum ----------------------------
#pragma unroll
    for (int j = 0; j < 8; ++j) {
        const int dcol = (dh + j) * 16 + m;
#pragma unroll
        for (int v = 0; v < 8; ++v) {
            int krow = kt * 16 + v + 8 * half;
            atomicAdd(&Vg[krow * D_DIM + dcol], vacc[j][v]);
        }
    }
    if (tid < K_DIM) atomicAdd(&asumg[tid], as_lds[tid]);
}

// ---------------------------------------------------------------------------
// Kernel 2: V = Vacc - c * a_sum ; intra-norm over K ; row L2 ; write y
// ---------------------------------------------------------------------------
__global__ void nv_finalize_kernel(const float* __restrict__ Vg,
                                   const float* __restrict__ asumg,
                                   const float* __restrict__ cmat,
                                   float* __restrict__ out) {
    extern __shared__ char smem[];
    float* Vs = (float*)smem;                       // 64*256 f32 = 64KB
    const int tid = threadIdx.x;
    const float EPS = 1e-12f;

    {   // thread d: build column, intra-normalize over K (axis 0)
        const int d = tid;
        float cn = 0.0f;
        for (int k = 0; k < K_DIM; ++k) {
            float v = Vg[k * D_DIM + d] - cmat[k * D_DIM + d] * asumg[k];
            Vs[k * D_DIM + d] = v;
            cn += v * v;
        }
        float inv = 1.0f / fmaxf(sqrtf(cn), EPS);
        for (int k = 0; k < K_DIM; ++k)
            Vs[k * D_DIM + d] *= inv;
    }
    __syncthreads();

    // thread k (<64): row L2 over D (axis 1), write output
    if (tid < K_DIM) {
        const int k = tid;
        float rn = 0.0f;
        for (int d = 0; d < D_DIM; ++d) {
            float v = Vs[k * D_DIM + d];
            rn += v * v;
        }
        float inv = 1.0f / fmaxf(sqrtf(rn), EPS);
        for (int d = 0; d < D_DIM; ++d)
            out[k * D_DIM + d] = Vs[k * D_DIM + d] * inv;
    }
}

// ---------------------------------------------------------------------------
extern "C" void kernel_launch(void* const* d_in, const int* in_sizes, int n_in,
                              void* d_out, int out_size, void* d_ws, size_t ws_size,
                              hipStream_t stream) {
    const float* xf     = (const float*)d_in[0];   // (1,256,512,512)
    const float* cmat   = (const float*)d_in[1];   // (64,256)
    const float* conv_w = (const float*)d_in[2];   // (64,256)
    const float* conv_b = (const float*)d_in[3];   // (64,)
    float* out = (float*)d_out;                    // (64,256)

    float* Vg    = (float*)d_ws;                   // 16384 f32
    float* asumg = Vg + K_DIM * D_DIM;             // 64 f32

    const int nz = K_DIM * D_DIM + K_DIM;
    nv_zero_kernel<<<(nz + 255) / 256, 256, 0, stream>>>(Vg, nz);

    nv_fused_kernel<<<NBLOCKS, NTHREADS, SM_TOT, stream>>>(
        xf, conv_w, conv_b, Vg, asumg);

    nv_finalize_kernel<<<1, 256, K_DIM * D_DIM * sizeof(float), stream>>>(
        Vg, asumg, cmat, out);
}
